// KernalAnsatz_65481071406993
// MI455X (gfx1250) — matrix operand order repaired and checked by reading, not compiled, not run on hardware
//
#include <hip/hip_runtime.h>

typedef __attribute__((ext_vector_type(16))) _Float16 v16h;
typedef __attribute__((ext_vector_type(8)))  _Float16 v8h;
typedef __attribute__((ext_vector_type(8)))  float    v8f;
typedef __attribute__((ext_vector_type(4)))  float    v4f;

// ---------------------------------------------------------------------------
// Phase A: fp32 row -> fp16 row + fp32 squared norm. One thread per 64-elem row.
// ---------------------------------------------------------------------------
__global__ void rbf_prep_kernel(const float* __restrict__ src,
                                _Float16* __restrict__ dst16,
                                float* __restrict__ norms,
                                int rows) {
    int r = blockIdx.x * blockDim.x + threadIdx.x;
    if (r >= rows) return;

    const v4f* p4 = reinterpret_cast<const v4f*>(src + (size_t)r * 64);
    _Float16 h[64];
    float acc = 0.0f;
#pragma unroll
    for (int c = 0; c < 16; ++c) {
        v4f v = p4[c];
#pragma unroll
        for (int e = 0; e < 4; ++e) {
            float f = v[e];
            acc += f * f;
            h[c * 4 + e] = (_Float16)f;
        }
    }
    v8h* q8 = reinterpret_cast<v8h*>(dst16 + (size_t)r * 64);
#pragma unroll
    for (int c = 0; c < 8; ++c) {
        v8h t;
#pragma unroll
        for (int e = 0; e < 8; ++e) t[e] = h[c * 8 + e];
        q8[c] = t;
    }
    norms[r] = acc;
}

// ---------------------------------------------------------------------------
// Phase B: one wave32 per 64x64 output tile. 4x4 grid of 16x16 WMMA tiles,
// K=64 as two K=32 steps -> 32 x v_wmma_f32_16x16x32_f16 per wave.
//
// Per-lane operand layouts (CDNA5 ISA 7.12.2, wave32):
//   A (16x32 f16, M x K): lane m (0-15) holds row m, K = {0..7, 16..23};
//                         lane m+16 holds row m, K = {8..15, 24..31}.
//   B (32x16 f16, K x N): lane n (0-15) holds col n (= row n of Y), K = {0..7, 16..23};
//                         lane n+16 holds K = {8..15, 24..31}.
//   C/D (16x16 f32): VGPR r, lanes 0-15 -> (M=r, N=lane); lanes 16-31 -> M=r+8.
// ---------------------------------------------------------------------------
__global__ __launch_bounds__(32)
void rbf_wmma_kernel(const _Float16* __restrict__ x16,
                     const _Float16* __restrict__ y16,
                     const float* __restrict__ x2,
                     const float* __restrict__ y2,
                     const float* __restrict__ gptr,
                     float* __restrict__ out,
                     int Mcols) {
    const int lane   = threadIdx.x;
    const int laneLo = lane & 15;
    const int laneHi = lane >> 4;      // 0 or 1
    const int nBase  = blockIdx.x * 64;
    const int mBase  = blockIdx.y * 64;

    v8f c[4][4] = {};

#pragma unroll
    for (int kb = 0; kb < 2; ++kb) {
        const int kOff = kb * 32;

        v16h a[4], b[4];
#pragma unroll
        for (int mt = 0; mt < 4; ++mt) {
            const _Float16* base =
                x16 + (size_t)(mBase + mt * 16 + laneLo) * 64 + kOff + laneHi * 8;
            v8h lo = *reinterpret_cast<const v8h*>(base);        // K chunk 0
            v8h hi = *reinterpret_cast<const v8h*>(base + 16);   // K chunk +16
            a[mt] = __builtin_shufflevector(lo, hi,
                                            0, 1, 2, 3, 4, 5, 6, 7,
                                            8, 9, 10, 11, 12, 13, 14, 15);
        }
#pragma unroll
        for (int nt = 0; nt < 4; ++nt) {
            const _Float16* base =
                y16 + (size_t)(nBase + nt * 16 + laneLo) * 64 + kOff + laneHi * 8;
            v8h lo = *reinterpret_cast<const v8h*>(base);        // K chunk 0
            v8h hi = *reinterpret_cast<const v8h*>(base + 16);   // K chunk +16
            b[nt] = __builtin_shufflevector(lo, hi,
                                            0, 1, 2, 3, 4, 5, 6, 7,
                                            8, 9, 10, 11, 12, 13, 14, 15);
        }

#pragma unroll
        for (int mt = 0; mt < 4; ++mt)
#pragma unroll
            for (int nt = 0; nt < 4; ++nt)
                c[mt][nt] = __builtin_amdgcn_wmma_f32_16x16x32_f16(
                    /*neg_a=*/false, a[mt], /*neg_b=*/false, b[nt],
                    /*c_mod=*/(short)0, c[mt][nt],
                    /*reuse_a=*/false, /*reuse_b=*/false);
    }

    // Epilogue: out[i,j] = exp(-gamma * (x2[i] + y2[j] - 2*c))
    const float gamma = gptr[0];

    v8f xs[4];
#pragma unroll
    for (int mt = 0; mt < 4; ++mt)
        xs[mt] = *reinterpret_cast<const v8f*>(x2 + mBase + mt * 16 + laneHi * 8);

#pragma unroll
    for (int nt = 0; nt < 4; ++nt) {
        const int j   = nBase + nt * 16 + laneLo;
        const float y2v = y2[j];
#pragma unroll
        for (int mt = 0; mt < 4; ++mt) {
            const int iBase = mBase + mt * 16 + laneHi * 8;
#pragma unroll
            for (int r = 0; r < 8; ++r) {
                const int i = iBase + r;
                float d2  = xs[mt][r] + y2v - 2.0f * c[mt][nt][r];
                float val = __expf(-gamma * d2);
                out[(size_t)i * Mcols + j] = val;
            }
        }
    }
}

// ---------------------------------------------------------------------------
// Launcher
// ---------------------------------------------------------------------------
extern "C" void kernel_launch(void* const* d_in, const int* in_sizes, int n_in,
                              void* d_out, int out_size, void* d_ws, size_t ws_size,
                              hipStream_t stream) {
    const float* x     = (const float*)d_in[0];
    const float* y     = (const float*)d_in[1];
    const float* gamma = (const float*)d_in[2];

    const int D  = 64;
    const int N  = in_sizes[0] / D;   // 8192 rows of x
    const int Mr = in_sizes[1] / D;   // 8192 rows of y

    // Workspace layout: [x16 | y16 | x2 | y2]
    char* ws = (char*)d_ws;
    _Float16* x16 = (_Float16*)ws;
    _Float16* y16 = (_Float16*)(ws + (size_t)N * D * sizeof(_Float16));
    float*    x2  = (float*)(ws + ((size_t)N + (size_t)Mr) * D * sizeof(_Float16));
    float*    y2  = x2 + N;

    // Phase A: convert + norms
    {
        int threads = 256;
        hipLaunchKernelGGL(rbf_prep_kernel, dim3((N + threads - 1) / threads),
                           dim3(threads), 0, stream, x, x16, x2, N);
        hipLaunchKernelGGL(rbf_prep_kernel, dim3((Mr + threads - 1) / threads),
                           dim3(threads), 0, stream, y, y16, y2, Mr);
    }

    // Phase B: WMMA GEMM + exp epilogue. One wave per 64x64 tile.
    {
        dim3 grid(Mr / 64, N / 64);
        dim3 block(32);
        hipLaunchKernelGGL(rbf_wmma_kernel, grid, block, 0, stream,
                           x16, y16, x2, y2, gamma, (float*)d_out, Mr);
    }
}
